// MultiheadChannelAttention_44246753084206
// MI455X (gfx1250) — matrix operand order repaired and checked by reading, not compiled, not run on hardware
//
#include <hip/hip_runtime.h>
#include <hip/hip_bf16.h>

typedef __attribute__((ext_vector_type(16))) __bf16 v16bf;
typedef __attribute__((ext_vector_type(8)))  float  v8f;
typedef __attribute__((ext_vector_type(4)))  float  f32x4;
typedef __attribute__((ext_vector_type(2)))  unsigned int u32x2;
typedef int v4i __attribute__((vector_size(16)));   // matches builtin param type

#define D_MODEL 1024
#define N_HEAD  8
#define D_X     64
#define HD      512            // N_HEAD * D_X
#define QKV_N   1536
#define SEQ_L   8192
#define BATCH   4
#define M_TOTAL (BATCH * SEQ_L)  // 32768
#define LN_EPS  1e-6f

#define BM 128
#define BN 64
#define BK 32
#define LDK 40   // padded K-stride in bf16 elements (40*2B = 80B, 16B-aligned rows)
#define L_SLICES 8

#if defined(__has_builtin)
#if __has_builtin(__builtin_amdgcn_global_load_async_to_lds_b128)
#define HAVE_ASYNC_LDS 1
#endif
#endif

__device__ __forceinline__ unsigned short f2bf(float f) {
  unsigned int u = __float_as_uint(f);
  u += 0x7FFFu + ((u >> 16) & 1u);      // round-to-nearest-even
  return (unsigned short)(u >> 16);
}

union FragU { v16bf v; f32x4 f4[2]; };

// Gather one WMMA operand fragment (two aligned b128 LDS loads).
__device__ __forceinline__ v16bf ldsFrag(const unsigned short* base, int off0, int off1) {
  FragU f;
  f.f4[0] = *reinterpret_cast<const f32x4*>(base + off0);
  f.f4[1] = *reinterpret_cast<const f32x4*>(base + off1);
  return f.v;
}

#ifdef HAVE_ASYNC_LDS
__device__ __forceinline__ void asyncCopy16(const void* g, void* l) {
  // inttoptr casts: always-legal conversion into AS1/AS3 pointers.
  // Low 32 bits of a generic LDS pointer are the LDS byte offset (ISA aperture rule).
  __builtin_amdgcn_global_load_async_to_lds_b128(
      (__attribute__((address_space(1))) v4i*)(unsigned long long)(uintptr_t)g,
      (__attribute__((address_space(3))) v4i*)(unsigned int)(uintptr_t)l, 0, 0);
}
#endif

// ---------------- Kernel 1: qkv = x @ w_qkv + b_qkv  (store bf16) ----------------
__device__ __forceinline__ void k1_load(const float* __restrict__ x, const float* __restrict__ w,
                                        int rowBase, int colBase, int kk, int tid,
                                        f32x4* aR, f32x4* bR) {
#pragma unroll
  for (int i = 0; i < 4; ++i) {
    int f = tid + i * 256;
    int m = f >> 3, k0 = (f & 7) * 4;
    aR[i] = *reinterpret_cast<const f32x4*>(x + (size_t)(rowBase + m) * D_MODEL + kk + k0);
  }
#pragma unroll
  for (int i = 0; i < 2; ++i) {
    int f = tid + i * 256;
    int kr = f >> 4, n0 = (f & 15) * 4;
    bR[i] = *reinterpret_cast<const f32x4*>(w + (size_t)(kk + kr) * QKV_N + colBase + n0);
  }
}

__device__ __forceinline__ void k1_store(unsigned short* As, unsigned short* Bs, int tid,
                                         const f32x4* aR, const f32x4* bR) {
#pragma unroll
  for (int i = 0; i < 4; ++i) {
    int f = tid + i * 256;
    int m = f >> 3, k0 = (f & 7) * 4;
    u32x2 pk;
    pk.x = (unsigned)f2bf(aR[i].x) | ((unsigned)f2bf(aR[i].y) << 16);
    pk.y = (unsigned)f2bf(aR[i].z) | ((unsigned)f2bf(aR[i].w) << 16);
    *reinterpret_cast<u32x2*>(As + m * LDK + k0) = pk;   // one ds_store_b64
  }
#pragma unroll
  for (int i = 0; i < 2; ++i) {
    int f = tid + i * 256;
    int kr = f >> 4, n0 = (f & 15) * 4;
    Bs[(n0 + 0) * LDK + kr] = f2bf(bR[i].x);
    Bs[(n0 + 1) * LDK + kr] = f2bf(bR[i].y);
    Bs[(n0 + 2) * LDK + kr] = f2bf(bR[i].z);
    Bs[(n0 + 3) * LDK + kr] = f2bf(bR[i].w);
  }
}

__global__ __launch_bounds__(256) void qkv_gemm_kernel(
    const float* __restrict__ x, const float* __restrict__ w,
    const float* __restrict__ bias, unsigned short* __restrict__ qkv) {
  __shared__ unsigned short As[2 * BM * LDK];
  __shared__ unsigned short Bs[2 * BN * LDK];
  const int tid = threadIdx.x;
  const int lane = tid & 31, wid = tid >> 5;
  const int waveM = wid & 3, waveN = wid >> 2;
  const int rowBase = blockIdx.x * BM;
  const int colBase = blockIdx.y * BN;

  f32x4 aR[4], bR[2];
  k1_load(x, w, rowBase, colBase, 0, tid, aR, bR);
  k1_store(As, Bs, tid, aR, bR);
  __syncthreads();

  v8f acc[2][2] = {};
  int p = 0;
  for (int kk = 0; kk < D_MODEL; kk += BK, p ^= 1) {
    const bool hasNext = (kk + BK) < D_MODEL;
    if (hasNext) k1_load(x, w, rowBase, colBase, kk + BK, tid, aR, bR);
    if (kk + 2 * BK < D_MODEL)
      __builtin_prefetch(x + (size_t)(rowBase + (tid >> 3)) * D_MODEL + kk + 2 * BK, 0, 1);

    const unsigned short* Ap = As + p * (BM * LDK);
    const unsigned short* Bp = Bs + p * (BN * LDK);
    const int k0a = (lane < 16) ? 0 : 8;
    const int k0b = (lane < 16) ? 0 : 8;
    v16bf aF[2], bF[2];
#pragma unroll
    for (int mt = 0; mt < 2; ++mt) {
      int r = waveM * 32 + mt * 16 + (lane & 15);
      aF[mt] = ldsFrag(Ap, r * LDK + k0a, r * LDK + k0a + 16);
    }
#pragma unroll
    for (int nt = 0; nt < 2; ++nt) {
      int n = waveN * 32 + nt * 16 + (lane & 15);
      bF[nt] = ldsFrag(Bp, n * LDK + k0b, n * LDK + k0b + 16);
    }
#pragma unroll
    for (int mt = 0; mt < 2; ++mt)
#pragma unroll
      for (int nt = 0; nt < 2; ++nt)
        acc[mt][nt] = __builtin_amdgcn_wmma_f32_16x16x32_bf16(
            false, aF[mt], false, bF[nt], (short)0, acc[mt][nt], false, false);

    if (hasNext) k1_store(As + (p ^ 1) * (BM * LDK), Bs + (p ^ 1) * (BN * LDK), tid, aR, bR);
    __syncthreads();
  }
#pragma unroll
  for (int mt = 0; mt < 2; ++mt)
#pragma unroll
    for (int nt = 0; nt < 2; ++nt) {
      int col = colBase + waveN * 32 + nt * 16 + (lane & 15);
      float bv = bias[col];
      int r0 = rowBase + waveM * 32 + mt * 16 + ((lane >> 4) ? 8 : 0);
#pragma unroll
      for (int i = 0; i < 8; ++i)
        qkv[(size_t)(r0 + i) * QKV_N + col] = f2bf(acc[mt][nt][i] + bv);
    }
}

// ---- Kernel 2a: partial scores = Q^T K over an L-slice, atomicAdd into ws ----
__global__ __launch_bounds__(256) void scores_kernel(
    const unsigned short* __restrict__ qkv, float* __restrict__ scoresG) {
  __shared__ unsigned short Qt[D_X * LDK];  // [d][l]
  __shared__ unsigned short Kt[D_X * LDK];  // [e][l]
  const int tid = threadIdx.x;
  const int lane = tid & 31, wid = tid >> 5;
  const int bh = blockIdx.x;
  const int b = bh >> 3, h = bh & 7;
  const size_t rowBase = (size_t)b * SEQ_L;
  const int lSliceBeg = blockIdx.y * (SEQ_L / L_SLICES);
  const int lSliceEnd = lSliceBeg + (SEQ_L / L_SLICES);
  const int qOff = h * D_X;
  const int kOff = HD + h * D_X;
  const int dTile  = wid >> 1;        // 16 (d,e) tiles, 2 per wave
  const int eTile0 = (wid & 1) * 2;

  v8f acc0 = {}, acc1 = {};
  for (int lc = lSliceBeg; lc < lSliceEnd; lc += 32) {
#pragma unroll
    for (int i = 0; i < 4; ++i) {
      int f = tid + i * 256;
      int l = f >> 5, d0 = (f & 31) * 2;
      size_t base = (rowBase + lc + l) * QKV_N;
      unsigned int qv = *reinterpret_cast<const unsigned int*>(qkv + base + qOff + d0);
      Qt[(d0 + 0) * LDK + l] = (unsigned short)(qv & 0xFFFFu);
      Qt[(d0 + 1) * LDK + l] = (unsigned short)(qv >> 16);
      unsigned int kv = *reinterpret_cast<const unsigned int*>(qkv + base + kOff + d0);
      Kt[(d0 + 0) * LDK + l] = (unsigned short)(kv & 0xFFFFu);
      Kt[(d0 + 1) * LDK + l] = (unsigned short)(kv >> 16);
    }
    if (lc + 32 < lSliceEnd)
      __builtin_prefetch(qkv + (rowBase + lc + 32 + (tid >> 5)) * QKV_N + qOff, 0, 1);
    __syncthreads();

    const int k0a = (lane < 16) ? 0 : 8;
    const int k0b = (lane < 16) ? 0 : 8;
    int r = dTile * 16 + (lane & 15);
    v16bf aF = ldsFrag(Qt, r * LDK + k0a, r * LDK + k0a + 16);
    int n0 = eTile0 * 16 + (lane & 15);
    v16bf bF0 = ldsFrag(Kt, n0 * LDK + k0b, n0 * LDK + k0b + 16);
    v16bf bF1 = ldsFrag(Kt, (n0 + 16) * LDK + k0b, (n0 + 16) * LDK + k0b + 16);
    acc0 = __builtin_amdgcn_wmma_f32_16x16x32_bf16(false, aF, false, bF0, (short)0, acc0, false, false);
    acc1 = __builtin_amdgcn_wmma_f32_16x16x32_bf16(false, aF, false, bF1, (short)0, acc1, false, false);
    __syncthreads();
  }
  float* Sg = scoresG + (size_t)bh * D_X * D_X;
  int d0 = dTile * 16 + ((lane >> 4) ? 8 : 0);
  int e0 = eTile0 * 16 + (lane & 15);
#pragma unroll
  for (int i = 0; i < 8; ++i) {
    atomicAdd(&Sg[(d0 + i) * D_X + e0],      acc0[i]);
    atomicAdd(&Sg[(d0 + i) * D_X + e0 + 16], acc1[i]);
  }
}

// ---- Kernel 2b: softmax(scores/8) then fold into w_fc -> w_mod (bf16) ----
__global__ __launch_bounds__(256) void softmax_fold_kernel(
    const float* __restrict__ scoresG, const float* __restrict__ w_fc,
    unsigned short* __restrict__ wmod) {
  __shared__ float S[D_X * D_X];
  const int tid = threadIdx.x;
  const int bh = blockIdx.x;
  const int b = bh >> 3, h = bh & 7;
  const float* Sg = scoresG + (size_t)bh * D_X * D_X;
#pragma unroll
  for (int i = 0; i < 16; ++i) S[tid + i * 256] = Sg[tid + i * 256];
  __syncthreads();
  if (tid < D_X) {
    float m = -1e30f;
    for (int e = 0; e < D_X; ++e) m = fmaxf(m, S[tid * D_X + e]);
    float s = 0.f;
    for (int e = 0; e < D_X; ++e) {
      float v = __expf((S[tid * D_X + e] - m) * 0.125f);
      S[tid * D_X + e] = v; s += v;
    }
    float inv = 1.f / s;
    for (int e = 0; e < D_X; ++e) S[tid * D_X + e] *= inv;
  }
  __syncthreads();
  // w_mod[h*64+d, j] = sum_e attn[d][e] * w_fc[h*64+e, j]
  for (int jj = 0; jj < 4; ++jj) {
    int j = tid + jj * 256;
    for (int dg = 0; dg < 8; ++dg) {
      float a[8] = {};
      for (int e = 0; e < D_X; ++e) {
        float wv = w_fc[(size_t)(h * D_X + e) * D_MODEL + j];
#pragma unroll
        for (int dd = 0; dd < 8; ++dd)
          a[dd] += S[(dg * 8 + dd) * D_X + e] * wv;
      }
#pragma unroll
      for (int dd = 0; dd < 8; ++dd)
        wmod[((size_t)b * HD + h * D_X + dg * 8 + dd) * D_MODEL + j] = f2bf(a[dd]);
    }
  }
}

// ---------------- Kernel 3: gout = V @ w_mod_b  (f32 out) ----------------
__device__ __forceinline__ void k3_loadB(const unsigned short* __restrict__ wB,
                                         int colBase, int kk, int tid, unsigned int* bR) {
#pragma unroll
  for (int i = 0; i < 4; ++i) {
    int f = tid + i * 256;
    int kr = f >> 5, n0 = (f & 31) * 2;
    bR[i] = *reinterpret_cast<const unsigned int*>(wB + (size_t)(kk + kr) * D_MODEL + colBase + n0);
  }
}
__device__ __forceinline__ void k3_storeB(unsigned short* Bs, int tid, const unsigned int* bR) {
#pragma unroll
  for (int i = 0; i < 4; ++i) {
    int f = tid + i * 256;
    int kr = f >> 5, n0 = (f & 31) * 2;
    Bs[(n0 + 0) * LDK + kr] = (unsigned short)(bR[i] & 0xFFFFu);
    Bs[(n0 + 1) * LDK + kr] = (unsigned short)(bR[i] >> 16);
  }
}
__device__ __forceinline__ void k3_stageA(const unsigned short* __restrict__ qkv,
                                          unsigned short* As, int rowBase, int kk, int tid
#ifndef HAVE_ASYNC_LDS
                                          , f32x4* aHold, bool doLoad
#endif
                                          ) {
#pragma unroll
  for (int i = 0; i < 2; ++i) {
    int f = tid + i * 256;
    int m = f >> 2, k0 = (f & 3) * 8;
    const unsigned short* gp = qkv + (size_t)(rowBase + m) * QKV_N + 2 * HD + kk + k0;
    unsigned short* lp = As + m * LDK + k0;
#ifdef HAVE_ASYNC_LDS
    asyncCopy16(gp, lp);                         // global_load_async_to_lds_b128
#else
    if (doLoad) aHold[i] = *reinterpret_cast<const f32x4*>(gp);
    else        *reinterpret_cast<f32x4*>(lp) = aHold[i];
#endif
  }
}

__global__ __launch_bounds__(256) void fc_gemm_kernel(
    const unsigned short* __restrict__ qkv, const unsigned short* __restrict__ wmod,
    float* __restrict__ gout) {
  __shared__ unsigned short As[2 * BM * LDK];
  __shared__ unsigned short Bs[2 * BN * LDK];
  const int tid = threadIdx.x;
  const int lane = tid & 31, wid = tid >> 5;
  const int waveM = wid & 3, waveN = wid >> 2;
  const int rowBase = blockIdx.x * BM;
  const int colBase = blockIdx.y * BN;
  const int b = rowBase / SEQ_L;   // BM divides SEQ_L
  const unsigned short* wB = wmod + (size_t)b * HD * D_MODEL;

  unsigned int bR[4];
#ifndef HAVE_ASYNC_LDS
  f32x4 aHold[2];
#endif
  // prologue: stage chunk 0 into buffer 0
#ifdef HAVE_ASYNC_LDS
  k3_stageA(qkv, As, rowBase, 0, tid);
#else
  k3_stageA(qkv, As, rowBase, 0, tid, aHold, true);
  k3_stageA(qkv, As, rowBase, 0, tid, aHold, false);
#endif
  k3_loadB(wB, colBase, 0, tid, bR);
  k3_storeB(Bs, tid, bR);
#ifdef HAVE_ASYNC_LDS
  asm volatile("s_wait_asynccnt 0x0" ::: "memory");
#endif
  __syncthreads();

  v8f acc[2][2] = {};
  int p = 0;
  for (int kk = 0; kk < HD; kk += BK, p ^= 1) {
    const bool hasNext = (kk + BK) < HD;
    if (hasNext) {
#ifdef HAVE_ASYNC_LDS
      k3_stageA(qkv, As + (p ^ 1) * (BM * LDK), rowBase, kk + BK, tid);
#else
      k3_stageA(qkv, As + (p ^ 1) * (BM * LDK), rowBase, kk + BK, tid, aHold, true);
#endif
      k3_loadB(wB, colBase, kk + BK, tid, bR);
    }
    const unsigned short* Ap = As + p * (BM * LDK);
    const unsigned short* Bp = Bs + p * (BN * LDK);
    const int k0a = (lane < 16) ? 0 : 8;
    const int k0b = (lane < 16) ? 0 : 8;
    v16bf aF[2], bF[2];
#pragma unroll
    for (int mt = 0; mt < 2; ++mt) {
      int r = waveM * 32 + mt * 16 + (lane & 15);
      aF[mt] = ldsFrag(Ap, r * LDK + k0a, r * LDK + k0a + 16);
    }
#pragma unroll
    for (int nt = 0; nt < 2; ++nt) {
      int n = waveN * 32 + nt * 16 + (lane & 15);
      bF[nt] = ldsFrag(Bp, n * LDK + k0b, n * LDK + k0b + 16);
    }
#pragma unroll
    for (int mt = 0; mt < 2; ++mt)
#pragma unroll
      for (int nt = 0; nt < 2; ++nt)
        acc[mt][nt] = __builtin_amdgcn_wmma_f32_16x16x32_bf16(
            false, aF[mt], false, bF[nt], (short)0, acc[mt][nt], false, false);

    if (hasNext) {
      k3_storeB(Bs + (p ^ 1) * (BN * LDK), tid, bR);
#ifndef HAVE_ASYNC_LDS
      k3_stageA(qkv, As + (p ^ 1) * (BM * LDK), rowBase, kk + BK, tid, aHold, false);
#endif
    }
#ifdef HAVE_ASYNC_LDS
    asm volatile("s_wait_asynccnt 0x0" ::: "memory");
#endif
    __syncthreads();
  }
#pragma unroll
  for (int mt = 0; mt < 2; ++mt)
#pragma unroll
    for (int nt = 0; nt < 2; ++nt) {
      int col = colBase + waveN * 32 + nt * 16 + (lane & 15);
      int r0 = rowBase + waveM * 32 + mt * 16 + ((lane >> 4) ? 8 : 0);
#pragma unroll
      for (int i = 0; i < 8; ++i)
        gout[(size_t)(r0 + i) * D_MODEL + col] = acc[mt][nt][i];
    }
}

// ------------- Kernel 4: out = LN(gout + b_fc + x) * gamma + beta -------------
__global__ __launch_bounds__(256) void ln_kernel(
    const float* __restrict__ gout, const float* __restrict__ x,
    const float* __restrict__ b_fc, const float* __restrict__ gamma,
    const float* __restrict__ beta, float* __restrict__ out) {
  __shared__ float red[16];
  const int tid = threadIdx.x;
  const size_t r = blockIdx.x;
  float v[4], s1 = 0.f, s2 = 0.f;
#pragma unroll
  for (int i = 0; i < 4; ++i) {
    int j = tid + i * 256;
    float val = gout[r * D_MODEL + j] + x[r * D_MODEL + j] + b_fc[j];
    v[i] = val; s1 += val; s2 += val * val;
  }
  for (int off = 16; off > 0; off >>= 1) {   // wave32 reduction
    s1 += __shfl_down(s1, off);
    s2 += __shfl_down(s2, off);
  }
  const int lane = tid & 31, wid = tid >> 5;
  if (lane == 0) { red[wid] = s1; red[8 + wid] = s2; }
  __syncthreads();
  if (tid == 0) {
    float a = 0.f, bsum = 0.f;
    for (int w = 0; w < 8; ++w) { a += red[w]; bsum += red[8 + w]; }
    red[0] = a; red[8] = bsum;
  }
  __syncthreads();
  float mu  = red[0] * (1.f / D_MODEL);
  float var = red[8] * (1.f / D_MODEL) - mu * mu;
  float rs  = rsqrtf(var + LN_EPS);
#pragma unroll
  for (int i = 0; i < 4; ++i) {
    int j = tid + i * 256;
    out[r * D_MODEL + j] = (v[i] - mu) * rs * gamma[j] + beta[j];
  }
}

extern "C" void kernel_launch(void* const* d_in, const int* in_sizes, int n_in,
                              void* d_out, int out_size, void* d_ws, size_t ws_size,
                              hipStream_t stream) {
  const float* x     = (const float*)d_in[0];
  const float* w_qkv = (const float*)d_in[1];
  const float* b_qkv = (const float*)d_in[2];
  const float* w_fc  = (const float*)d_in[3];
  const float* b_fc  = (const float*)d_in[4];
  const float* gamma = (const float*)d_in[5];
  const float* beta  = (const float*)d_in[6];
  float* out = (float*)d_out;

  // workspace layout
  char* ws = (char*)d_ws;
  unsigned short* qkv = (unsigned short*)ws;                 // 32768*1536 bf16  ~100.7 MB
  ws += (size_t)M_TOTAL * QKV_N * 2;
  unsigned short* wmod = (unsigned short*)ws;                // 4*512*1024 bf16  ~4 MB
  ws += (size_t)BATCH * HD * D_MODEL * 2;
  float* scoresG = (float*)ws;                               // 32*64*64 f32     512 KB
  size_t scoresBytes = (size_t)BATCH * N_HEAD * D_X * D_X * 4;
  ws += scoresBytes;
  float* gout = (float*)ws;                                  // 32768*1024 f32   ~134 MB

  qkv_gemm_kernel<<<dim3(M_TOTAL / BM, QKV_N / BN), 256, 0, stream>>>(x, w_qkv, b_qkv, qkv);
  (void)hipMemsetAsync(scoresG, 0, scoresBytes, stream);
  scores_kernel<<<dim3(BATCH * N_HEAD, L_SLICES), 256, 0, stream>>>(qkv, scoresG);
  softmax_fold_kernel<<<dim3(BATCH * N_HEAD), 256, 0, stream>>>(scoresG, w_fc, wmod);
  fc_gemm_kernel<<<dim3(M_TOTAL / BM, D_MODEL / BN), 256, 0, stream>>>(qkv, wmod, gout);
  ln_kernel<<<dim3(M_TOTAL), 256, 0, stream>>>(gout, x, b_fc, gamma, beta, out);
}
